// SkinAwareMoEAdapter_79886391705738
// MI455X (gfx1250) — compile-verified, matrix operand order checked by reading, not run on hardware
//
#include <hip/hip_runtime.h>
#include <hip/hip_bf16.h>

// ---------------- problem constants (match reference) ----------------
#define E_   8
#define K_   2
#define N_   16384
#define H_   2048
#define B_   128
#define EPS_ 1e-6f

typedef __bf16 bf16;
typedef __attribute__((ext_vector_type(16))) __bf16 bf16x16;
typedef __attribute__((ext_vector_type(8)))  __bf16 bf16x8;
typedef __attribute__((ext_vector_type(8)))  float  f32x8;

// ---------------------------------------------------------------------
// 0) zero the aux accumulators (16 floats) each call
// ---------------------------------------------------------------------
__global__ void zero_accum_kernel(float* __restrict__ acc) {
    if (threadIdx.x < 16) acc[threadIdx.x] = 0.0f;
}

// ---------------------------------------------------------------------
// 1) fp32 -> bf16 conversion (RNE), 4 elements/thread, packed b64 store
// ---------------------------------------------------------------------
__global__ void cvt_bf16_kernel(const float* __restrict__ src,
                                bf16* __restrict__ dst, size_t n4) {
    size_t i = (size_t)blockIdx.x * blockDim.x + threadIdx.x;
    if (i < n4) {
        const float4 v = ((const float4*)src)[i];
        union { bf16 b[4]; uint2 u; } o;
        o.b[0] = (bf16)v.x; o.b[1] = (bf16)v.y;
        o.b[2] = (bf16)v.z; o.b[3] = (bf16)v.w;
        ((uint2*)dst)[i] = o.u;
    }
}

// ---------------------------------------------------------------------
// 2) router: one wave (32 lanes) per token
// ---------------------------------------------------------------------
__global__ __launch_bounds__(256)
void router_kernel(const float* __restrict__ x,
                   const float* __restrict__ skin,
                   const float* __restrict__ Wimg,
                   const float* __restrict__ Wskin,
                   float* __restrict__ wout,
                   float* __restrict__ probSum,
                   float* __restrict__ freqCnt) {
    __shared__ float blkProb[E_];
    __shared__ float blkFreq[E_];
    const int lane = threadIdx.x & 31;
    const int wave = threadIdx.x >> 5;
    if (threadIdx.x < E_) { blkProb[threadIdx.x] = 0.f; blkFreq[threadIdx.x] = 0.f; }
    __syncthreads();

    const int wavesPerBlk = blockDim.x >> 5;
    const int gwave  = blockIdx.x * wavesPerBlk + wave;
    const int nwaves = gridDim.x * wavesPerBlk;

    for (int t = gwave; t < N_; t += nwaves) {
        float part[E_];
#pragma unroll
        for (int e = 0; e < E_; e++) part[e] = 0.f;
        const float* xr = x + (size_t)t * H_;
        for (int h = lane; h < H_; h += 32) {
            float xv = xr[h];
#pragma unroll
            for (int e = 0; e < E_; e++)
                part[e] = fmaf(xv, Wimg[e * H_ + h], part[e]);
        }
#pragma unroll
        for (int e = 0; e < E_; e++)
            for (int off = 16; off; off >>= 1)
                part[e] += __shfl_xor(part[e], off, 32);

        if (lane == 0) {
            const float s0 = skin[t * 3 + 0], s1 = skin[t * 3 + 1], s2 = skin[t * 3 + 2];
            float lg[E_]; float mx = -1e30f;
#pragma unroll
            for (int e = 0; e < E_; e++) {
                lg[e] = part[e] + s0 * Wskin[e * 3 + 0]
                                + s1 * Wskin[e * 3 + 1]
                                + s2 * Wskin[e * 3 + 2];
                mx = fmaxf(mx, lg[e]);
            }
            float p[E_]; float den = 0.f;
#pragma unroll
            for (int e = 0; e < E_; e++) { p[e] = __expf(lg[e] - mx); den += p[e]; }
            const float inv = 1.f / den;
#pragma unroll
            for (int e = 0; e < E_; e++) p[e] *= inv;
            // top-2 (first index wins ties, like jax top_k)
            int i1 = 0; float v1 = p[0];
#pragma unroll
            for (int e = 1; e < E_; e++) if (p[e] > v1) { v1 = p[e]; i1 = e; }
            int i2 = -1; float v2 = -1.f;
#pragma unroll
            for (int e = 0; e < E_; e++) if (e != i1 && p[e] > v2) { v2 = p[e]; i2 = e; }
            const float inv2 = 1.f / (v1 + v2 + EPS_);
#pragma unroll
            for (int e = 0; e < E_; e++) {
                float wv = (e == i1) ? v1 * inv2 : ((e == i2) ? v2 * inv2 : 0.f);
                wout[(size_t)t * E_ + e] = wv;
                atomicAdd(&blkProb[e], p[e]);
            }
            atomicAdd(&blkFreq[i1], 1.f);
            atomicAdd(&blkFreq[i2], 1.f);
        }
    }
    __syncthreads();
    if (threadIdx.x < E_) {
        atomicAdd(&probSum[threadIdx.x], blkProb[threadIdx.x]);
        atomicAdd(&freqCnt[threadIdx.x], blkFreq[threadIdx.x]);
    }
}

// ---------------------------------------------------------------------
// fragment union: CDNA5 16-bit A/B layout — lane = row/col (lane&15),
// K values { klo..klo+7, 16+klo..23+klo } with klo = (lane>>4)*8.
// ---------------------------------------------------------------------
union FragBF { bf16x16 v; bf16x8 h[2]; };

#define XS_STRIDE 2056   // 2048 + 8: 16B-aligned, +4 banks/row -> conflict-free
#define HW_STRIDE 136    // 128 + 8:  16B-aligned, +4 banks/row -> conflict-free

// ---------------------------------------------------------------------
// 3) MoE expert kernel: one block per 16-token tile, 128 threads (4 waves)
// ---------------------------------------------------------------------
__global__ __launch_bounds__(128)
void moe_kernel(const float* __restrict__ x,
                const bf16*  __restrict__ xb,
                const bf16*  __restrict__ W1b,
                const float* __restrict__ b1,
                const bf16*  __restrict__ W2b,
                const float* __restrict__ b2,
                const float* __restrict__ wN,
                float* __restrict__ out) {
    __shared__ __align__(16) bf16 Xs[16][XS_STRIDE];   // staged x tile (bf16)
    __shared__ __align__(16) bf16 Hw[E_][16][HW_STRIDE];
    __shared__ float tileW[16][E_];
    __shared__ int   eflag[E_];

    const int tile = blockIdx.x;           // 0..N/16-1
    const int tok0 = tile * 16;
    const int lane = threadIdx.x & 31;
    const int wave = threadIdx.x >> 5;
    const int col  = lane & 15;            // fragment row/col index
    const int klo  = (lane >> 4) * 8;      // per-lane K offset
    const int hi   = lane >> 4;

    // ---- async-stage the x tile (16 x 2048 bf16 = 64KB) into LDS -----
    // 4096 chunks of 16B; 128 threads x 32 iterations; ASYNCcnt path.
    {
        const bf16* gbase = xb + (size_t)tok0 * H_;
#pragma unroll 4
        for (int it = 0; it < 32; ++it) {
            int i = threadIdx.x + 128 * it;
            int r = i >> 8;                 // 256 chunks per row
            int c = i & 255;                // 16B chunk within row
            const bf16* g = gbase + (size_t)r * H_ + c * 8;
            unsigned lds  = (unsigned)(uintptr_t)(&Xs[r][c * 8]);
            asm volatile("global_load_async_to_lds_b128 %0, %1, off"
                         :: "v"(lds), "v"(g) : "memory");
        }
    }

    // combine weights while the DMA is in flight
    if (threadIdx.x < 16 * E_) {
        int m = threadIdx.x >> 3, e = threadIdx.x & 7;
        tileW[m][e] = wN[(size_t)(tok0 + m) * E_ + e];
    }
    __syncthreads();
    if (threadIdx.x < E_) {
        int f = 0;
        for (int m = 0; m < 16; m++) f |= (tileW[m][threadIdx.x] != 0.f);
        eflag[threadIdx.x] = f;
    }
    // wait for this wave's async transfers, then barrier for cross-wave vis
    asm volatile("s_wait_asynccnt 0x0" ::: "memory");
    __syncthreads();

    // ---------------- stage 1: Hw[e] = w * relu(X @ W1_e^T + b1_e) ----
    const int ct0 = wave * 2;              // each wave owns 2 b-column tiles
    for (int e = 0; e < E_; e++) {
        if (!eflag[e]) continue;           // block-uniform branch
        f32x8 acc0 = {0.f,0.f,0.f,0.f,0.f,0.f,0.f,0.f};
        f32x8 acc1 = {0.f,0.f,0.f,0.f,0.f,0.f,0.f,0.f};
        const bf16* arow  = &Xs[col][klo];                       // LDS
        const bf16* brow0 = W1b + ((size_t)e * B_ + ct0 * 16 + col) * H_ + klo;
        const bf16* brow1 = brow0 + (size_t)16 * H_;
#pragma unroll 2
        for (int k0 = 0; k0 < H_; k0 += 32) {
            FragBF a, b0, b1f;
            a.h[0]   = *(const bf16x8*)(arow  + k0);
            a.h[1]   = *(const bf16x8*)(arow  + k0 + 16);
            b0.h[0]  = *(const bf16x8*)(brow0 + k0);
            b0.h[1]  = *(const bf16x8*)(brow0 + k0 + 16);
            b1f.h[0] = *(const bf16x8*)(brow1 + k0);
            b1f.h[1] = *(const bf16x8*)(brow1 + k0 + 16);
            acc0 = __builtin_amdgcn_wmma_f32_16x16x32_bf16(
                false, a.v, false, b0.v, (short)0, acc0, false, false);
            acc1 = __builtin_amdgcn_wmma_f32_16x16x32_bf16(
                false, a.v, false, b1f.v, (short)0, acc1, false, false);
        }
        const float bb0 = b1[e * B_ + ct0 * 16 + col];
        const float bb1 = b1[e * B_ + (ct0 + 1) * 16 + col];
#pragma unroll
        for (int r = 0; r < 8; r++) {
            int m = r + hi * 8;
            float wv = tileW[m][e];
            float h0 = fmaxf(acc0[r] + bb0, 0.f) * wv;
            float h1 = fmaxf(acc1[r] + bb1, 0.f) * wv;
            Hw[e][m][ct0 * 16 + col]       = (bf16)h0;
            Hw[e][m][(ct0 + 1) * 16 + col] = (bf16)h1;
        }
    }
    __syncthreads();

    // ---------------- stage 2: out = x + sum_e Hw_e @ W2_e^T + w@b2 ---
    for (int j = 0; j < 32; j++) {
        const int ct   = j * 4 + wave;     // 128 column tiles over 4 waves
        const int hcol = ct * 16 + col;
        f32x8 acc = {0.f,0.f,0.f,0.f,0.f,0.f,0.f,0.f};
        for (int e = 0; e < E_; e++) {
            if (!eflag[e]) continue;
            const bf16* arow = &Hw[e][col][klo];
            const bf16* brow = W2b + ((size_t)e * H_ + hcol) * B_ + klo;
#pragma unroll
            for (int k0 = 0; k0 < B_; k0 += 32) {
                FragBF a, b;
                a.h[0] = *(const bf16x8*)(arow + k0);
                a.h[1] = *(const bf16x8*)(arow + k0 + 16);
                b.h[0] = *(const bf16x8*)(brow + k0);
                b.h[1] = *(const bf16x8*)(brow + k0 + 16);
                acc = __builtin_amdgcn_wmma_f32_16x16x32_bf16(
                    false, a.v, false, b.v, (short)0, acc, false, false);
            }
        }
        float b2v[E_];
#pragma unroll
        for (int e = 0; e < E_; e++) b2v[e] = b2[(size_t)e * H_ + hcol];
#pragma unroll
        for (int r = 0; r < 8; r++) {
            int m = r + hi * 8;
            float s = 0.f;
#pragma unroll
            for (int e = 0; e < E_; e++) s = fmaf(tileW[m][e], b2v[e], s);
            size_t idx = (size_t)(tok0 + m) * H_ + hcol;
            out[idx] = x[idx] + acc[r] + s;
        }
    }
}

// ---------------------------------------------------------------------
// 4) finalize aux loss -> d_out[N*H]
// ---------------------------------------------------------------------
__global__ void finalize_kernel(const float* __restrict__ probSum,
                                const float* __restrict__ freqCnt,
                                float* __restrict__ out_aux) {
    if (threadIdx.x == 0 && blockIdx.x == 0) {
        float s = 0.f;
        for (int e = 0; e < E_; e++)
            s += (probSum[e] / (float)N_) * (freqCnt[e] / (float)N_);
        *out_aux = s * (float)E_;
    }
}

// ---------------------------------------------------------------------
extern "C" void kernel_launch(void* const* d_in, const int* in_sizes, int n_in,
                              void* d_out, int out_size, void* d_ws, size_t ws_size,
                              hipStream_t stream) {
    const float* x     = (const float*)d_in[0];
    const float* skin  = (const float*)d_in[1];
    const float* Wimg  = (const float*)d_in[2];
    const float* Wskin = (const float*)d_in[3];
    const float* W1    = (const float*)d_in[4];
    const float* b1    = (const float*)d_in[5];
    const float* W2    = (const float*)d_in[6];
    const float* b2    = (const float*)d_in[7];
    float* out = (float*)d_out;

    // workspace layout
    char* ws = (char*)d_ws;
    float* wN      = (float*)ws;                                   // N*E f32 (512KB)
    float* probSum = (float*)(ws + (size_t)N_ * E_ * sizeof(float));
    float* freqCnt = probSum + 8;
    bf16*  xb  = (bf16*)(ws + (1ull << 20));                       // N*H bf16 (64MB)
    bf16*  W1b = xb  + (size_t)N_ * H_;                            // E*B*H bf16 (4MB)
    bf16*  W2b = W1b + (size_t)E_ * B_ * H_;                       // E*H*B bf16 (4MB)

    // 0) zero aux accumulators
    zero_accum_kernel<<<1, 32, 0, stream>>>(probSum);

    // 1) bf16 conversions
    {
        size_t n4 = (size_t)N_ * H_ / 4;
        cvt_bf16_kernel<<<(unsigned)((n4 + 255) / 256), 256, 0, stream>>>(x, xb, n4);
        size_t w4 = (size_t)E_ * B_ * H_ / 4;
        cvt_bf16_kernel<<<(unsigned)((w4 + 255) / 256), 256, 0, stream>>>(W1, W1b, w4);
        cvt_bf16_kernel<<<(unsigned)((w4 + 255) / 256), 256, 0, stream>>>(W2, W2b, w4);
    }

    // 2) router (2048 waves, 8 tokens each)
    router_kernel<<<256, 256, 0, stream>>>(x, skin, Wimg, Wskin, wN, probSum, freqCnt);

    // 3) expert GEMMs + combine + residual
    moe_kernel<<<N_ / 16, 128, 0, stream>>>(x, xb, W1b, b1, W2b, b2, wN, out);

    // 4) aux loss scalar
    finalize_kernel<<<1, 32, 0, stream>>>(probSum, freqCnt, out + (size_t)N_ * H_);
}
